// MultiHeadSelfAttention_83571473645575
// MI455X (gfx1250) — matrix operand order repaired and checked by reading, not compiled, not run on hardware
//
#include <hip/hip_runtime.h>
#include <hip/hip_bf16.h>
#include <stddef.h>

// ---------------------------------------------------------------------------
// MHA forward, MI455X (gfx1250, wave32, WMMA f16 -> f32 accumulate)
//   B=2, S=2048, D=1024, H=16, hd=64
// Pipeline:
//   0a) cast x fp32 -> f16                      (xh    [4096,1024])
//   0b) transpose+cast w_qkv  -> f16 Bt layout  (wqkvT [3072,1024])
//   0c) transpose+cast w_proj -> f16 Bt layout  (wprojT[1024,1024])
//   1)  QKV GEMM (WMMA, double-buffered K-pipe) -> Q*0.125, K, V^T  f16
//   2)  flash attention (WMMA + online softmax, P via LDS, hoisted V loads)
//   3)  out-proj GEMM (WMMA, double-buffered K-pipe) + bias -> fp32 d_out
// Workspace: 48 MB of f16 tensors laid out below.
// ---------------------------------------------------------------------------

typedef __attribute__((ext_vector_type(16))) _Float16 v16h;
typedef __attribute__((ext_vector_type(8)))  _Float16 v8h;
typedef __attribute__((ext_vector_type(8)))  float    v8f;

union Frag16 { v16h v; v8h h[2]; };

#define S_LEN 2048
#define DMODEL 1024
#define NHEADS 16
#define HD 64
#define MTOT 4096          // B*S

__device__ __forceinline__ void load_afrag(Frag16& f, const _Float16* p) {
    f.h[0] = *(const v8h*)(p);          // K = halfbase + 0..7
    f.h[1] = *(const v8h*)(p + 16);     // K = halfbase + 16..23
}
__device__ __forceinline__ void load_bfrag(Frag16& f, const _Float16* p) {
    f.h[0] = *(const v8h*)(p);          // K = halfbase + 0..7
    f.h[1] = *(const v8h*)(p + 16);     // K = halfbase + 16..23
}

// ---------------------------------------------------------------- cast fp32->f16
__global__ __launch_bounds__(256) void cast_f16_kernel(const float* __restrict__ x,
                                                       _Float16* __restrict__ y, int n4) {
    int i = blockIdx.x * blockDim.x + threadIdx.x;
    if (i < n4) {
        float4 v = reinterpret_cast<const float4*>(x)[i];
        _Float16 o0 = (_Float16)v.x, o1 = (_Float16)v.y;
        _Float16 o2 = (_Float16)v.z, o3 = (_Float16)v.w;
        _Float16* p = y + (size_t)i * 4;
        p[0] = o0; p[1] = o1; p[2] = o2; p[3] = o3;
    }
}

// --------------------------------------------- transpose+cast: W[K,N] -> Wt[N,K] f16
__global__ __launch_bounds__(256) void transpose_cast_kernel(const float* __restrict__ W,
                                                             _Float16* __restrict__ Wt,
                                                             int K, int N) {
    __shared__ float tile[32][33];
    int tx = threadIdx.x & 31;
    int ty = threadIdx.x >> 5;          // 0..7
    int bn = blockIdx.x * 32;
    int bk = blockIdx.y * 32;
#pragma unroll
    for (int j = 0; j < 32; j += 8)
        tile[ty + j][tx] = W[(size_t)(bk + ty + j) * N + bn + tx];
    __syncthreads();
#pragma unroll
    for (int j = 0; j < 32; j += 8)
        Wt[(size_t)(bn + ty + j) * K + bk + tx] = (_Float16)tile[tx][ty + j];
}

// ---------------------------------------------------------------- GEMM core
// 16 rows x 64 cols per wave, K contraction in double-buffered steps of 32.
// aptr = Ah + (m0+l16)*K + half*8 ; bp[t] = Bt + (n0+t*16+l16)*K + half*8.
__device__ __forceinline__ void wmma_gemm_16x64(const _Float16* __restrict__ aptr,
                                                const _Float16* __restrict__ bp0,
                                                int K, v8f acc[4]) {
    const _Float16* bp[4];
#pragma unroll
    for (int t = 0; t < 4; ++t) bp[t] = bp0 + (size_t)t * 16 * K;

    Frag16 af0, af1, bf0[4], bf1[4];
    load_afrag(af0, aptr);
#pragma unroll
    for (int t = 0; t < 4; ++t) load_bfrag(bf0[t], bp[t]);

    for (int k0 = 0; k0 < K; k0 += 64) {
        // stage k0+32 while computing k0
        load_afrag(af1, aptr + k0 + 32);
#pragma unroll
        for (int t = 0; t < 4; ++t) load_bfrag(bf1[t], bp[t] + k0 + 32);
        __builtin_prefetch(aptr + k0 + 64, 0, 1);
#pragma unroll
        for (int t = 0; t < 4; ++t)
            acc[t] = __builtin_amdgcn_wmma_f32_16x16x32_f16(
                false, af0.v, false, bf0[t].v, (short)0, acc[t], false, false);
        // stage k0+64 while computing k0+32
        if (k0 + 64 < K) {
            load_afrag(af0, aptr + k0 + 64);
#pragma unroll
            for (int t = 0; t < 4; ++t) load_bfrag(bf0[t], bp[t] + k0 + 64);
        }
#pragma unroll
        for (int t = 0; t < 4; ++t)
            acc[t] = __builtin_amdgcn_wmma_f32_16x16x32_f16(
                false, af1.v, false, bf1[t].v, (short)0, acc[t], false, false);
    }
}

// ------------------------------------------------------------------ QKV GEMM
// C[4096,3072] = Ah[4096,1024] @ Bt^T. 8 waves/block (4M x 2N). Grid (24, 64).
__global__ __launch_bounds__(256) void gemm_qkv_kernel(const _Float16* __restrict__ Ah,
                                                       const _Float16* __restrict__ Bt,
                                                       const float* __restrict__ bias,
                                                       _Float16* __restrict__ Qh,
                                                       _Float16* __restrict__ Kh,
                                                       _Float16* __restrict__ Vt) {
    const int K = DMODEL;
    int lane = threadIdx.x & 31;
    int w    = threadIdx.x >> 5;
    int l16  = lane & 15;
    int half = lane >> 4;
    int wm = w & 3, wn = w >> 2;
    int m0 = blockIdx.y * 64 + wm * 16;
    int n0 = blockIdx.x * 128 + wn * 64;

    v8f acc[4] = {};
    wmma_gemm_16x64(Ah + (size_t)(m0 + l16) * K + half * 8,
                    Bt + (size_t)(n0 + l16) * K + half * 8, K, acc);

    // Epilogue: scatter into Q (scaled 1/sqrt(hd)), K, V^T as f16.
#pragma unroll
    for (int t = 0; t < 4; ++t) {
        int c = n0 + t * 16 + l16;          // 0..3071
        float bv = bias[c];
        int which = c >> 10;                // 0=Q 1=K 2=V
        int rem = c & 1023;
        int h = rem >> 6, d = rem & 63;
#pragma unroll
        for (int r = 0; r < 8; ++r) {
            int m = m0 + half * 8 + r;      // 0..4095
            int b = m >> 11, sl = m & 2047;
            int bh = b * NHEADS + h;
            float val = acc[t][r] + bv;
            if (which == 0) {
                Qh[((size_t)bh * S_LEN + sl) * HD + d] = (_Float16)(val * 0.125f);
            } else if (which == 1) {
                Kh[((size_t)bh * S_LEN + sl) * HD + d] = (_Float16)val;
            } else {
                Vt[((size_t)bh * HD + d) * S_LEN + sl] = (_Float16)val;
            }
        }
    }
}

// ----------------------------------------------------------- flash attention
// Grid (S/64, B*H), 128 threads = 4 independent waves; wave owns 16 query rows.
__global__ __launch_bounds__(128) void attn_kernel(const _Float16* __restrict__ Qh,
                                                   const _Float16* __restrict__ Kh,
                                                   const _Float16* __restrict__ Vt,
                                                   _Float16* __restrict__ attn) {
    __shared__ alignas(32) _Float16 pbuf[4][16 * 32];   // per-wave P tile
    int lane = threadIdx.x & 31;
    int w    = threadIdx.x >> 5;
    int l16  = lane & 15;
    int half = lane >> 4;
    int bh = blockIdx.y;
    int q0 = blockIdx.x * 64 + w * 16;

    const _Float16* Qb = Qh + (size_t)bh * S_LEN * HD;
    const _Float16* Kb = Kh + (size_t)bh * S_LEN * HD;
    const _Float16* Vb = Vt + (size_t)bh * HD * S_LEN;
    _Float16* pw = &pbuf[w][0];

    // Resident Q fragments (hd=64 -> two K-steps of 32).
    Frag16 qa[2];
#pragma unroll
    for (int kk = 0; kk < 2; ++kk)
        load_afrag(qa[kk], Qb + (size_t)(q0 + l16) * HD + kk * 32 + half * 8);

    v8f acc[4] = {};
    float mrow[8], lrow[8];
#pragma unroll
    for (int r = 0; r < 8; ++r) { mrow[r] = -1e30f; lrow[r] = 0.0f; }

    int nch = (q0 + 47) >> 5;               // causal: chunks of 32 keys
    for (int kc = 0; kc < nch; ++kc) {
        int kb = kc * 32;
        // ---- scores: 2 key tiles x 2 K-steps
        v8f s[2];
#pragma unroll
        for (int t = 0; t < 2; ++t) {
            v8f c = {};
#pragma unroll
            for (int kk = 0; kk < 2; ++kk) {
                Frag16 bf;
                load_bfrag(bf, Kb + (size_t)(kb + t * 16 + l16) * HD + kk * 32 + half * 8);
                c = __builtin_amdgcn_wmma_f32_16x16x32_f16(
                    false, qa[kk].v, false, bf.v, (short)0, c, false, false);
            }
            s[t] = c;
        }
        // ---- hoist V fragment loads: independent of softmax, so their latency
        //      is covered by ~30 VALU softmax ops + the LDS P round-trip below.
        Frag16 vf[4];
#pragma unroll
        for (int t = 0; t < 4; ++t)
            load_bfrag(vf[t], Vb + (size_t)(t * 16 + l16) * S_LEN + kb + half * 8);
        // ---- causal mask + per-row chunk max (16-lane reduce inside each half)
        float chmax[8];
#pragma unroll
        for (int r = 0; r < 8; ++r) {
            int row = q0 + half * 8 + r;
            float a0 = (kb + l16      <= row) ? s[0][r] : -1e30f;
            float a1 = (kb + 16 + l16 <= row) ? s[1][r] : -1e30f;
            s[0][r] = a0; s[1][r] = a1;
            float mx = fmaxf(a0, a1);
#pragma unroll
            for (int off = 1; off < 16; off <<= 1)
                mx = fmaxf(mx, __shfl_xor(mx, off, 32));
            chmax[r] = mx;
        }
        // ---- online softmax: rescale, exp, row sums, P -> LDS (A-layout source)
#pragma unroll
        for (int r = 0; r < 8; ++r) {
            float mnew = fmaxf(mrow[r], chmax[r]);
            float corr = __expf(mrow[r] - mnew);
            mrow[r] = mnew;
            float p0 = __expf(s[0][r] - mnew);
            float p1 = __expf(s[1][r] - mnew);
            float rs = p0 + p1;
#pragma unroll
            for (int off = 1; off < 16; off <<= 1)
                rs += __shfl_xor(rs, off, 32);
            lrow[r] = lrow[r] * corr + rs;
#pragma unroll
            for (int t = 0; t < 4; ++t) acc[t][r] *= corr;
            pw[(half * 8 + r) * 32 + l16]      = (_Float16)p0;
            pw[(half * 8 + r) * 32 + 16 + l16] = (_Float16)p1;
        }
        // ---- P as A-fragment (wave-private LDS, DScnt keeps ordering)
        Frag16 pa;
        load_afrag(pa, pw + l16 * 32 + half * 8);
        // ---- acc += P @ V over 4 hd tiles (V^T rows are contiguous in keys)
#pragma unroll
        for (int t = 0; t < 4; ++t)
            acc[t] = __builtin_amdgcn_wmma_f32_16x16x32_f16(
                false, pa.v, false, vf[t].v, (short)0, acc[t], false, false);
    }
    // ---- epilogue: 1/l, store f16 into [B,S,D] (heads re-interleaved)
    int b = bh >> 4, h = bh & 15;
#pragma unroll
    for (int r = 0; r < 8; ++r) {
        float inv = 1.0f / lrow[r];
        int srow = q0 + half * 8 + r;
        size_t base = ((size_t)b * S_LEN + srow) * DMODEL + h * HD;
#pragma unroll
        for (int t = 0; t < 4; ++t)
            attn[base + t * 16 + l16] = (_Float16)(acc[t][r] * inv);
    }
}

// --------------------------------------------------------------- out-proj GEMM
// out[4096,1024] = attn[4096,1024] @ w_proj + b_proj, fp32 output. Grid (8, 64).
__global__ __launch_bounds__(256) void gemm_proj_kernel(const _Float16* __restrict__ Ah,
                                                        const _Float16* __restrict__ Bt,
                                                        const float* __restrict__ bias,
                                                        float* __restrict__ out) {
    const int K = DMODEL;
    int lane = threadIdx.x & 31;
    int w    = threadIdx.x >> 5;
    int l16  = lane & 15;
    int half = lane >> 4;
    int wm = w & 3, wn = w >> 2;
    int m0 = blockIdx.y * 64 + wm * 16;
    int n0 = blockIdx.x * 128 + wn * 64;

    v8f acc[4] = {};
    wmma_gemm_16x64(Ah + (size_t)(m0 + l16) * K + half * 8,
                    Bt + (size_t)(n0 + l16) * K + half * 8, K, acc);

#pragma unroll
    for (int t = 0; t < 4; ++t) {
        int c = n0 + t * 16 + l16;
        float bv = bias[c];
#pragma unroll
        for (int r = 0; r < 8; ++r) {
            int m = m0 + half * 8 + r;
            out[(size_t)m * DMODEL + c] = acc[t][r] + bv;
        }
    }
}

// ---------------------------------------------------------------------------
extern "C" void kernel_launch(void* const* d_in, const int* in_sizes, int n_in,
                              void* d_out, int out_size, void* d_ws, size_t ws_size,
                              hipStream_t stream) {
    const float* x      = (const float*)d_in[0];   // [2,2048,1024]
    const float* w_qkv  = (const float*)d_in[1];   // [1024,3072]
    const float* b_qkv  = (const float*)d_in[2];   // [3072]
    const float* w_proj = (const float*)d_in[3];   // [1024,1024]
    const float* b_proj = (const float*)d_in[4];   // [1024]
    float* out = (float*)d_out;

    // Workspace layout (f16), 48 MB total.
    char* ws = (char*)d_ws;
    _Float16* xh     = (_Float16*)(ws);                       // 4096*1024   = 8 MB
    _Float16* wqkvT  = (_Float16*)(ws + (8u   << 20));        // 3072*1024   = 6 MB
    _Float16* wprojT = (_Float16*)(ws + (14u  << 20));        // 1024*1024   = 2 MB
    _Float16* Qh     = (_Float16*)(ws + (16u  << 20));        // [2,16,2048,64] = 8 MB
    _Float16* Kh     = (_Float16*)(ws + (24u  << 20));        // 8 MB
    _Float16* Vt     = (_Float16*)(ws + (32u  << 20));        // [2,16,64,2048] = 8 MB
    _Float16* attn   = (_Float16*)(ws + (40u  << 20));        // [4096,1024] = 8 MB

    // 0a) x -> f16
    cast_f16_kernel<<<(MTOT * DMODEL / 4 + 255) / 256, 256, 0, stream>>>(x, xh, MTOT * DMODEL / 4);
    // 0b/0c) weights -> transposed f16
    transpose_cast_kernel<<<dim3(3 * DMODEL / 32, DMODEL / 32), 256, 0, stream>>>(w_qkv, wqkvT, DMODEL, 3 * DMODEL);
    transpose_cast_kernel<<<dim3(DMODEL / 32, DMODEL / 32), 256, 0, stream>>>(w_proj, wprojT, DMODEL, DMODEL);
    // 1) QKV projection
    gemm_qkv_kernel<<<dim3(3 * DMODEL / 128, MTOT / 64), 256, 0, stream>>>(xh, wqkvT, b_qkv, Qh, Kh, Vt);
    // 2) flash attention
    attn_kernel<<<dim3(S_LEN / 64, 2 * NHEADS), 128, 0, stream>>>(Qh, Kh, Vt, attn);
    // 3) output projection (+bias) -> fp32
    gemm_proj_kernel<<<dim3(DMODEL / 128, MTOT / 64), 256, 0, stream>>>(attn, wprojT, b_proj, out);
}
